// PhysNetInteraction_32289564131698
// MI455X (gfx1250) — matrix operand order repaired and checked by reading, not compiled, not run on hardware
//
#include <hip/hip_runtime.h>

// ---- shapes (fixed by the reference) ----
#define BB 4
#define AA 1024
#define NN 64
#define FF 128
#define RBASIS 25
#define M_TOT (BB*AA)       // 4096 rows
#define CUTOFF_F 5.0f

typedef __attribute__((ext_vector_type(16))) __bf16          v16bf;
typedef __attribute__((ext_vector_type(16))) unsigned short  v16u;
typedef __attribute__((ext_vector_type(8)))  unsigned short  v8u;
typedef __attribute__((ext_vector_type(8)))  float           v8f;

__device__ __forceinline__ unsigned short f2bf(float f) {
    union { float f; unsigned u; } v; v.f = f;
    unsigned r = v.u + 0x7FFFu + ((v.u >> 16) & 1u);   // round-to-nearest-even
    return (unsigned short)(r >> 16);
}

// swish = x * sigmoid(x); denominator >= 1 so fast v_rcp_f32 is safe
// (avoids the v_div_scale_f32 IEEE-division sequence on the critical path)
__device__ __forceinline__ float swishf(float x) {
    return x * __builtin_amdgcn_rcpf(1.0f + __expf(-x));
}

// ---------------------------------------------------------------------------
// prep: convert 9 FxF weight matrices + padded filt_w to bf16, pre-swizzled
// into a B-operand layout with contiguous k order (A uses the same order): for kt (K/32), ft (F/16):
//   dst[((kt*8 + ft)*32 + lane)*16 + j] = W[kt*32 + (lane>>4)*16 + j][ft*16 + (lane&15)]
// ---------------------------------------------------------------------------
__global__ __launch_bounds__(256) void prep_kernel(
    const float* __restrict__ m0, const float* __restrict__ m1,
    const float* __restrict__ m2, const float* __restrict__ m3,
    const float* __restrict__ m4, const float* __restrict__ m5,
    const float* __restrict__ m6, const float* __restrict__ m7,
    const float* __restrict__ m8, const float* __restrict__ filt,
    unsigned short* __restrict__ dst)
{
    int idx = blockIdx.x * 256 + threadIdx.x;
    const int WELEM = FF * FF;              // 16384
    if (idx < 9 * WELEM) {
        const float* mats[9] = {m0, m1, m2, m3, m4, m5, m6, m7, m8};
        int mat = idx >> 14;
        int e   = idx & (WELEM - 1);
        int k = e >> 7, f = e & 127;
        int kt = k >> 5, ft = f >> 4;
        int lane = (((k >> 4) & 1) << 4) | (f & 15);
        int j = k & 15;
        dst[mat * WELEM + ((kt * 8 + ft) * 32 + lane) * 16 + j] =
            f2bf(mats[mat][k * FF + f]);
    } else if (idx < 9 * WELEM + 32 * FF) {
        int e = idx - 9 * WELEM;            // over 32 x 128 (K padded 25->32)
        int k = e >> 7, f = e & 127;
        int ft = f >> 4;
        int lane = (((k >> 4) & 1) << 4) | (f & 15);
        int j = k & 15;
        float val = (k < RBASIS) ? filt[k * FF + f] : 0.0f;
        dst[9 * WELEM + (ft * 32 + lane) * 16 + j] = f2bf(val);
    }
}

// ---------------------------------------------------------------------------
// One 16x16 f32 tile of  act(16x128 bf16, LDS) @ W(128x128, pre-swizzled bf16)
// ---------------------------------------------------------------------------
__device__ __forceinline__ v8f gemm_tile(
    const unsigned short (*act)[FF],
    const unsigned short* __restrict__ wsw, int ftile, int lane)
{
    v8f acc = {};
    int arow  = lane & 15;
    int khalf = lane >> 4;
    #pragma unroll
    for (int kt = 0; kt < 4; ++kt) {
        int k0 = kt * 32;
        // A fragment: contiguous k order (k = 16*khalf + s), matching B swizzle
        v8u lo = *(const v8u*)&act[arow][k0 + khalf * 16];
        v8u hi = *(const v8u*)&act[arow][k0 + khalf * 16 + 8];
        v16u au;
        #pragma unroll
        for (int j = 0; j < 8; ++j) { au[j] = lo[j]; au[8 + j] = hi[j]; }
        // B fragment: one aligned 32B load from pre-swizzled weights
        v16u bu = *(const v16u*)(wsw + ((kt * 8 + ftile) * 32 + lane) * 16);
        v16bf a = __builtin_bit_cast(v16bf, au);
        v16bf b = __builtin_bit_cast(v16bf, bu);
        acc = __builtin_amdgcn_wmma_f32_16x16x32_bf16(
            false, a, false, b, (short)0, acc, false, false);
    }
    return acc;
}

// ---------------------------------------------------------------------------
// branch(x) = swish(residual(x)) @ dw + db, fused per 16-row block.
// 8 waves; wave w owns output column tile [16w, 16w+16).
// ---------------------------------------------------------------------------
__global__ __launch_bounds__(256) void branch_kernel(
    const float* __restrict__ xin,
    const unsigned short* __restrict__ w1s,
    const unsigned short* __restrict__ w2s,
    const unsigned short* __restrict__ dws,
    const float* __restrict__ b1, const float* __restrict__ b2,
    const float* __restrict__ db, float* __restrict__ out)
{
    __shared__ float          xrow[16][FF];
    __shared__ unsigned short act[16][FF];

    int tid = threadIdx.x;
    int lane = tid & 31, wv = tid >> 5;
    int rowbase = blockIdx.x * 16;

    // stage x rows, act = bf16(swish(x))
    for (int i = tid; i < 16 * FF; i += 256) {
        int r = i >> 7, c = i & 127;
        float v = xin[(size_t)(rowbase + r) * FF + c];
        xrow[r][c] = v;
        act[r][c]  = f2bf(swishf(v));
    }
    __syncthreads();

    int col = lane & 15, khalf = lane >> 4;
    int crow0 = khalf * 8;
    int ftile = wv;

    // GEMM1: h1 = swish(x) @ w1 + b1 ; act = bf16(swish(h1))
    v8f acc = gemm_tile(act, w1s, ftile, lane);
    float bias = b1[ftile * 16 + col];
    __syncthreads();
    #pragma unroll
    for (int i = 0; i < 8; ++i)
        act[crow0 + i][ftile * 16 + col] = f2bf(swishf(acc[i] + bias));
    __syncthreads();

    // GEMM2: h2 = act @ w2 + b2 ; r = x + h2 ; act = bf16(swish(r))
    acc = gemm_tile(act, w2s, ftile, lane);
    bias = b2[ftile * 16 + col];
    __syncthreads();
    #pragma unroll
    for (int i = 0; i < 8; ++i) {
        float r = xrow[crow0 + i][ftile * 16 + col] + acc[i] + bias;
        act[crow0 + i][ftile * 16 + col] = f2bf(swishf(r));
    }
    __syncthreads();

    // GEMM3: out = act @ dw + db
    acc = gemm_tile(act, dws, ftile, lane);
    bias = db[ftile * 16 + col];
    #pragma unroll
    for (int i = 0; i < 8; ++i)
        out[(size_t)(rowbase + crow0 + i) * FF + ftile * 16 + col] = acc[i] + bias;
}

// ---------------------------------------------------------------------------
// cfconv: one block per atom (b,a).
//   W = (f_ij @ filt_w) * mollifier(r)*mask   (WMMA, K padded 25->32)
//   conv[f] = sum_n W[n,f] * x_j[neighbor[n], f] ;  v = x_i + conv
// ---------------------------------------------------------------------------
__global__ __launch_bounds__(256) void cfconv_kernel(
    const float* __restrict__ x_i, const float* __restrict__ x_j,
    const float* __restrict__ r_ij, const int* __restrict__ neighbors,
    const float* __restrict__ nmask, const float* __restrict__ f_ij,
    const unsigned short* __restrict__ filt_s, float* __restrict__ vout)
{
    __shared__ float Wlds[NN][FF];
    __shared__ float cc[NN];
    __shared__ int   nbl[NN];
    __shared__ float part[2][FF];

    int m = blockIdx.x;              // 0..4095 = b*A + a
    int batch = m >> 10;             // A = 1024
    int tid = threadIdx.x, lane = tid & 31, wv = tid >> 5;

    const float* fij = f_ij + (size_t)m * NN * RBASIS;

    // prefetch this atom's 6.4KB f_ij slab (global_prefetch_b8) so the
    // scattered per-element bf16-conversion loads below hit cache
    if (tid < 50)
        __builtin_prefetch(fij + tid * 32, 0, 1);

    if (tid < NN) {
        float r = r_ij[(size_t)m * NN + tid];
        float d = r * (1.0f / CUTOFF_F);
        float safe = fminf(d, 1.0f - 1e-6f);
        float cv = (d < 1.0f) ? __expf(1.0f - 1.0f / (1.0f - safe * safe)) : 0.0f;
        cc[tid]  = cv * nmask[(size_t)m * NN + tid];
        nbl[tid] = neighbors[(size_t)m * NN + tid];
    }
    __syncthreads();

    int khalf = lane >> 4, col = lane & 15, arow = lane & 15;

    // B fragment (filt_w, ftile = wv) reused for all 4 n-tiles
    v16u bu = *(const v16u*)(filt_s + (wv * 32 + lane) * 16);
    v16bf bfrag = __builtin_bit_cast(v16bf, bu);

    #pragma unroll
    for (int ntile = 0; ntile < 4; ++ntile) {
        int n = ntile * 16 + arow;
        v16u au;
        #pragma unroll
        for (int j = 0; j < 8; ++j) {
            int k = khalf * 16 + j;
            au[j] = (k < RBASIS) ? f2bf(fij[n * RBASIS + k]) : (unsigned short)0;
        }
        #pragma unroll
        for (int j = 0; j < 8; ++j) {
            int k = khalf * 16 + 8 + j;
            au[8 + j] = (k < RBASIS) ? f2bf(fij[n * RBASIS + k]) : (unsigned short)0;
        }
        v16bf a = __builtin_bit_cast(v16bf, au);
        v8f acc = {};
        acc = __builtin_amdgcn_wmma_f32_16x16x32_bf16(
            false, a, false, bfrag, (short)0, acc, false, false);
        #pragma unroll
        for (int i = 0; i < 8; ++i) {
            int nn = ntile * 16 + i + 8 * khalf;
            Wlds[nn][wv * 16 + col] = acc[i] * cc[nn];
        }
    }
    __syncthreads();

    // gather + reduce over neighbors (x_j rows are L2-resident)
    int f = tid & 127, half = tid >> 7;
    const float* xjb = x_j + (size_t)batch * AA * FF;
    float s = 0.0f;
    #pragma unroll 4
    for (int n = half * 32; n < half * 32 + 32; ++n)
        s += Wlds[n][f] * xjb[(size_t)nbl[n] * FF + f];
    part[half][f] = s;
    __syncthreads();

    if (tid < FF)
        vout[(size_t)m * FF + tid] = x_i[(size_t)m * FF + tid] + part[0][tid] + part[1][tid];
}

// ---------------------------------------------------------------------------
extern "C" void kernel_launch(void* const* d_in, const int* in_sizes, int n_in,
                              void* d_out, int out_size, void* d_ws, size_t ws_size,
                              hipStream_t stream)
{
    const float* x         = (const float*)d_in[0];
    const float* r_ij      = (const float*)d_in[1];
    const int*   neighbors = (const int*)  d_in[2];
    const float* nmask     = (const float*)d_in[3];
    const float* f_ij      = (const float*)d_in[4];
    const float* filt_w    = (const float*)d_in[5];

    const float* bi_w1 = (const float*)d_in[6];  const float* bi_b1 = (const float*)d_in[7];
    const float* bi_w2 = (const float*)d_in[8];  const float* bi_b2 = (const float*)d_in[9];
    const float* bi_dw = (const float*)d_in[10]; const float* bi_db = (const float*)d_in[11];
    const float* bj_w1 = (const float*)d_in[12]; const float* bj_b1 = (const float*)d_in[13];
    const float* bj_w2 = (const float*)d_in[14]; const float* bj_b2 = (const float*)d_in[15];
    const float* bj_dw = (const float*)d_in[16]; const float* bj_db = (const float*)d_in[17];
    const float* bv_w1 = (const float*)d_in[18]; const float* bv_b1 = (const float*)d_in[19];
    const float* bv_w2 = (const float*)d_in[20]; const float* bv_b2 = (const float*)d_in[21];
    const float* bv_dw = (const float*)d_in[22]; const float* bv_db = (const float*)d_in[23];

    // workspace layout
    const int WELEM = FF * FF;                        // 16384
    const int NSWZ  = 9 * WELEM + 32 * FF;            // 151552 ushorts
    unsigned char* ws = (unsigned char*)d_ws;
    unsigned short* wsw   = (unsigned short*)ws;      // swizzled bf16 weights
    unsigned short* filts = wsw + 9 * WELEM;
    float* x_ib = (float*)(ws + (size_t)NSWZ * 2);    // 303104 B, 32B aligned
    float* x_jb = x_ib + (size_t)M_TOT * FF;
    float* vbuf = x_jb + (size_t)M_TOT * FF;

    prep_kernel<<<(NSWZ + 255) / 256, 256, 0, stream>>>(
        bi_w1, bi_w2, bi_dw, bj_w1, bj_w2, bj_dw, bv_w1, bv_w2, bv_dw,
        filt_w, wsw);

    branch_kernel<<<M_TOT / 16, 256, 0, stream>>>(
        x, wsw + 0 * WELEM, wsw + 1 * WELEM, wsw + 2 * WELEM,
        bi_b1, bi_b2, bi_db, x_ib);

    branch_kernel<<<M_TOT / 16, 256, 0, stream>>>(
        x, wsw + 3 * WELEM, wsw + 4 * WELEM, wsw + 5 * WELEM,
        bj_b1, bj_b2, bj_db, x_jb);

    cfconv_kernel<<<M_TOT, 256, 0, stream>>>(
        x_ib, x_jb, r_ij, neighbors, nmask, f_ij, filts, vbuf);

    branch_kernel<<<M_TOT / 16, 256, 0, stream>>>(
        vbuf, wsw + 6 * WELEM, wsw + 7 * WELEM, wsw + 8 * WELEM,
        bv_b1, bv_b2, bv_db, (float*)d_out);
}
